// PoseRNN_73229192397174
// MI455X (gfx1250) — matrix operand order repaired and checked by reading, not compiled, not run on hardware
//
#include <hip/hip_runtime.h>
#include <hip/hip_bf16.h>

// ---------------------------------------------------------------------------
// PoseRNN for MI455X (gfx1250): bf16 WMMA GEMMs (2x2 register-blocked, sized
// to avoid VGPR spills) + persistent grid-synchronized
// GRU recurrence.
// ---------------------------------------------------------------------------

typedef __bf16 bf16;
typedef __attribute__((ext_vector_type(16))) __bf16 v16bf;
typedef __attribute__((ext_vector_type(8)))  __bf16 v8bf;
typedef __attribute__((ext_vector_type(8)))  float  v8f;

#define H    768
#define H3   2304
#define BB   64
#define SS   256
#define BS   (BB * SS)   // 16384
#define RH   128         // regressor hidden

// ---- workspace layout (all offsets 256B aligned) ----
static const size_t OFF_GI  = 0;                                   // f32 [BS,2304]
static const size_t OFF_XBF = OFF_GI  + (size_t)BS * H3 * 4;       // bf16 [BS,768]
static const size_t OFF_YBF = OFF_XBF + (size_t)BS * H  * 2;       // bf16 [BS,768]
static const size_t OFF_WIH = OFF_YBF + (size_t)BS * H  * 2;       // bf16 [2,2304,768]
static const size_t OFF_WHH = OFF_WIH + (size_t)2  * H3 * H * 2;   // bf16 [2,2304,768]
static const size_t OFF_W1  = OFF_WHH + (size_t)2  * H3 * H * 2;   // bf16 [128,768]
static const size_t OFF_H   = OFF_W1  + (size_t)RH * H  * 2;       // f32  [64,768]
static const size_t OFF_HBF = OFF_H   + (size_t)BB * H  * 4;       // bf16 [64,768]
static const size_t OFF_GH  = OFF_HBF + (size_t)BB * H  * 2;       // f32  [64,2304]
static const size_t OFF_HDN = OFF_GH  + (size_t)BB * H3 * 4;       // f32  [BS,128]
static const size_t OFF_BAR = OFF_HDN + (size_t)BS * RH * 4;       // 2 x u32 barrier state

// ---------------------------------------------------------------------------
// WMMA fragment helpers (wave32, 16x16x32 bf16, f32 accumulate).
// A (16x32, MxK): lane l<16 holds row M=l, K = k0..k0+15;
//                 lane l+16 holds row M=l, K = k0+16..k0+31.
// B (32x16, KxN) from row-major W[N,K]: lane n<16 -> col N=n, K=k0..k0+15;
//                 lane n+16 -> col N=n, K=k0+16..k0+31  (32 contiguous bytes/lane).
// C/D (16x16 f32): lane l<16 VGPR v = (M=v, N=l); lane l+16 VGPR v = (M=v+8, N=l-16).
// ---------------------------------------------------------------------------
__device__ inline v16bf load_a16x32(const bf16* A, int lda, int m0, int k0, int lane) {
    const bf16* p = A + (size_t)(m0 + (lane & 15)) * lda + k0 + ((lane >> 4) << 4);
    union { v16bf v; v8bf h[2]; } u;
    u.h[0] = *(const v8bf*)(p);
    u.h[1] = *(const v8bf*)(p + 8);
    return u.v;
}

__device__ inline v16bf load_bT16x32(const bf16* W, int ldw, int n0, int k0, int lane) {
    const bf16* p = W + (size_t)(n0 + (lane & 15)) * ldw + k0 + ((lane >> 4) << 4);
    return *(const v16bf*)p;
}

// Register-blocked macro tile: D[MB][NB] += A(MB*16 x K) * W^T(K x NB*16).
// MB=NB=2 keeps acc+frags+pipelined operands well under 256 VGPRs (no spill).
template <int MB, int NB>
__device__ inline void wmma_macro_tile(const bf16* A, int lda, const bf16* W, int ldw,
                                       int m0, int n0, int K, int lane, v8f (&acc)[MB][NB]) {
    for (int k0 = 0; k0 < K; k0 += 32) {
        v16bf a[MB], b[NB];
#pragma unroll
        for (int i = 0; i < MB; ++i) a[i] = load_a16x32(A, lda, m0 + 16 * i, k0, lane);
#pragma unroll
        for (int j = 0; j < NB; ++j) b[j] = load_bT16x32(W, ldw, n0 + 16 * j, k0, lane);
        if (k0 + 32 < K) {
            __builtin_prefetch(A + (size_t)(m0 + (lane & 15)) * lda + k0 + 32, 0, 3);
            __builtin_prefetch(W + (size_t)(n0 + (lane & 15)) * ldw + k0 + 32, 0, 3);
        }
#pragma unroll
        for (int i = 0; i < MB; ++i)
#pragma unroll
            for (int j = 0; j < NB; ++j)
                acc[i][j] = __builtin_amdgcn_wmma_f32_16x16x32_bf16(
                    false, a[i], false, b[j], (short)0, acc[i][j], false, false);
    }
}

// Single-tile version (used on the recurrent critical path: parallelism > reuse).
__device__ inline v8f wmma_tile_bf16(const bf16* A, int lda, const bf16* W, int ldw,
                                     int m0, int n0, int K, int lane) {
    v8f acc = {0.f, 0.f, 0.f, 0.f, 0.f, 0.f, 0.f, 0.f};
    for (int k0 = 0; k0 < K; k0 += 32) {
        v16bf a = load_a16x32(A, lda, m0, k0, lane);
        v16bf b = load_bT16x32(W, ldw, n0, k0, lane);
        acc = __builtin_amdgcn_wmma_f32_16x16x32_bf16(false, a, false, b,
                                                      (short)0, acc, false, false);
    }
    return acc;
}

__device__ inline void store_tile_f32(float* C, int ldc, int m0, int n0, int lane, v8f acc) {
    int n  = n0 + (lane & 15);
    int mb = m0 + ((lane >> 4) << 3);
#pragma unroll
    for (int v = 0; v < 8; ++v) C[(size_t)(mb + v) * ldc + n] = acc[v];
}

// ---------------------------------------------------------------------------
// Device-wide barrier (generation counter). All blocks resident (64 blocks).
// ---------------------------------------------------------------------------
__device__ inline void grid_sync(unsigned* cnt, unsigned* gen, unsigned nblocks) {
    __threadfence();
    __syncthreads();
    if (threadIdx.x == 0) {
        unsigned g = __atomic_load_n(gen, __ATOMIC_RELAXED);
        if (atomicInc(cnt, nblocks - 1u) == nblocks - 1u) {
            __atomic_fetch_add(gen, 1u, __ATOMIC_RELEASE);
        } else {
            while (__atomic_load_n(gen, __ATOMIC_ACQUIRE) == g)
                __builtin_amdgcn_s_sleep(1);
        }
    }
    __syncthreads();
    __threadfence();
}

// ---------------------------------------------------------------------------
// Kernel 1: concat fv|fi -> bf16 X, convert weights to bf16, reset barrier.
// ---------------------------------------------------------------------------
__global__ void prep_kernel(const float* __restrict__ fv, const float* __restrict__ fi,
                            const float* __restrict__ W_ih, const float* __restrict__ W_hh,
                            const float* __restrict__ W1,
                            bf16* __restrict__ xbf, bf16* __restrict__ wihbf,
                            bf16* __restrict__ whhbf, bf16* __restrict__ w1bf,
                            unsigned* __restrict__ bar) {
    size_t tid = (size_t)blockIdx.x * blockDim.x + threadIdx.x;
    size_t nt  = (size_t)gridDim.x * blockDim.x;
    for (size_t i = tid; i < (size_t)BS * H; i += nt) {
        size_t c = i % H, r = i / H;
        float v = (c < 512) ? fv[r * 512 + c] : fi[r * 256 + (c - 512)];
        xbf[i] = (bf16)v;
    }
    for (size_t i = tid; i < (size_t)2 * H3 * H; i += nt) {
        wihbf[i] = (bf16)W_ih[i];
        whhbf[i] = (bf16)W_hh[i];
    }
    for (size_t i = tid; i < (size_t)RH * H; i += nt) w1bf[i] = (bf16)W1[i];
    if (tid < 2) bar[tid] = 0u;
}

// ---------------------------------------------------------------------------
// Kernel 2: GI = X @ W_ih^T for all B*S rows (off the recurrent critical path).
// M=16384, N=2304, K=768. 32x32 macro-tile per wave (4 WMMA per K-slice).
// ---------------------------------------------------------------------------
__global__ void gemm_gi_kernel(const bf16* __restrict__ X, const bf16* __restrict__ Wih,
                               float* __restrict__ GI) {
    const int MBK = 2, NBK = 2;
    int lane   = threadIdx.x & 31;
    int wid    = blockIdx.x * (blockDim.x >> 5) + (threadIdx.x >> 5);
    int nwaves = gridDim.x * (blockDim.x >> 5);
    const int MT = BS / (16 * MBK);  // 512
    const int NT = H3 / (16 * NBK);  // 72
    for (int job = wid; job < MT * NT; job += nwaves) {
        int m0 = (job / NT) * 16 * MBK;
        int n0 = (job % NT) * 16 * NBK;
        v8f acc[MBK][NBK];
#pragma unroll
        for (int i = 0; i < MBK; ++i)
#pragma unroll
            for (int j = 0; j < NBK; ++j)
                acc[i][j] = (v8f){0.f, 0.f, 0.f, 0.f, 0.f, 0.f, 0.f, 0.f};
        wmma_macro_tile<MBK, NBK>(X, H, Wih, H, m0, n0, H, lane, acc);
#pragma unroll
        for (int i = 0; i < MBK; ++i)
#pragma unroll
            for (int j = 0; j < NBK; ++j)
                store_tile_f32(GI, H3, m0 + 16 * i, n0 + 16 * j, lane, acc[i][j]);
    }
}

// ---------------------------------------------------------------------------
// Kernel 3: persistent GRU recurrence for one layer.
// Per step: phase1 GH = h_bf16 @ W_hh^T (4x144 WMMA tiles), grid barrier,
//           phase2 f32 gate math -> h, h_bf16, Y[t]; final h -> h_n output.
// ---------------------------------------------------------------------------
__global__ void gru_seq_kernel(const float* __restrict__ GI, const bf16* __restrict__ Whh,
                               const float* __restrict__ bih, const float* __restrict__ bhh,
                               float* __restrict__ h, bf16* __restrict__ hbf,
                               float* __restrict__ GH, bf16* __restrict__ Y,
                               float* __restrict__ hn_out,
                               unsigned* bar_cnt, unsigned* bar_gen, unsigned nblocks) {
    int lane    = threadIdx.x & 31;
    int wid     = blockIdx.x * (blockDim.x >> 5) + (threadIdx.x >> 5);
    int nwaves  = gridDim.x * (blockDim.x >> 5);
    size_t tid  = (size_t)blockIdx.x * blockDim.x + threadIdx.x;
    size_t nthr = (size_t)gridDim.x * blockDim.x;

    // h0 = 0
    for (size_t i = tid; i < (size_t)BB * H; i += nthr) { h[i] = 0.f; hbf[i] = (bf16)0.f; }
    grid_sync(bar_cnt, bar_gen, nblocks);

    const int NT = H3 / 16;  // 144 col tiles, 4 row tiles
    for (int t = 0; t < SS; ++t) {
        // ---- phase 1: GH = hbf @ Whh^T ----
        for (int tt = wid; tt < 4 * NT; tt += nwaves) {
            int mt = tt / NT, nt = tt % NT;
            v8f acc = wmma_tile_bf16(hbf, H, Whh, H, mt * 16, nt * 16, H, lane);
            store_tile_f32(GH, H3, mt * 16, nt * 16, lane, acc);
        }
        grid_sync(bar_cnt, bar_gen, nblocks);

        // ---- phase 2: gates + state update ----
        for (size_t i = tid; i < (size_t)BB * H; i += nthr) {
            int m = (int)(i / H), c = (int)(i % H);
            size_t gib = ((size_t)m * SS + t) * H3;
            size_t ghb = (size_t)m * H3;
            float ir  = GI[gib + c]         + bih[c];
            float iz  = GI[gib + H + c]     + bih[H + c];
            float inn = GI[gib + 2 * H + c] + bih[2 * H + c];
            float hr  = GH[ghb + c]         + bhh[c];
            float hz  = GH[ghb + H + c]     + bhh[H + c];
            float hn  = GH[ghb + 2 * H + c] + bhh[2 * H + c];
            float r = 1.f / (1.f + __expf(-(ir + hr)));
            float z = 1.f / (1.f + __expf(-(iz + hz)));
            float n = tanhf(inn + r * hn);
            float hp = h[i];
            float hv = (1.f - z) * n + z * hp;
            h[i]   = hv;
            hbf[i] = (bf16)hv;
            Y[((size_t)m * SS + t) * H + c] = (bf16)hv;
            if (t == SS - 1) hn_out[i] = hv;
        }
        grid_sync(bar_cnt, bar_gen, nblocks);
    }
}

// ---------------------------------------------------------------------------
// Kernel 4: HDN = LeakyReLU(Y1 @ W1^T + b1), WMMA (16384 x 128 x 768),
// 32x32 macro-tile per wave.
// ---------------------------------------------------------------------------
__global__ void regr1_kernel(const bf16* __restrict__ Y1, const bf16* __restrict__ W1bf,
                             const float* __restrict__ b1, float* __restrict__ HDN) {
    const int MBK = 2, NBK = 2;
    int lane   = threadIdx.x & 31;
    int wid    = blockIdx.x * (blockDim.x >> 5) + (threadIdx.x >> 5);
    int nwaves = gridDim.x * (blockDim.x >> 5);
    const int MT = BS / (16 * MBK);  // 512
    const int NT = RH / (16 * NBK);  // 4
    for (int job = wid; job < MT * NT; job += nwaves) {
        int m0 = (job / NT) * 16 * MBK;
        int n0 = (job % NT) * 16 * NBK;
        v8f acc[MBK][NBK];
#pragma unroll
        for (int i = 0; i < MBK; ++i)
#pragma unroll
            for (int j = 0; j < NBK; ++j)
                acc[i][j] = (v8f){0.f, 0.f, 0.f, 0.f, 0.f, 0.f, 0.f, 0.f};
        wmma_macro_tile<MBK, NBK>(Y1, H, W1bf, H, m0, n0, H, lane, acc);
#pragma unroll
        for (int i = 0; i < MBK; ++i) {
#pragma unroll
            for (int j = 0; j < NBK; ++j) {
                int n  = n0 + 16 * j + (lane & 15);
                int mb = m0 + 16 * i + ((lane >> 4) << 3);
                float bv = b1[n];
#pragma unroll
                for (int v = 0; v < 8; ++v) {
                    float x = acc[i][j][v] + bv;
                    HDN[(size_t)(mb + v) * RH + n] = (x >= 0.f) ? x : 0.1f * x;
                }
            }
        }
    }
}

// ---------------------------------------------------------------------------
// Kernel 5: pose = HDN @ W2^T + b2  (N=6, tiny: scalar f32).
// ---------------------------------------------------------------------------
__global__ void regr2_kernel(const float* __restrict__ HDN, const float* __restrict__ W2,
                             const float* __restrict__ b2, float* __restrict__ pose) {
    size_t nt = (size_t)gridDim.x * blockDim.x;
    for (size_t r = (size_t)blockIdx.x * blockDim.x + threadIdx.x; r < BS; r += nt) {
        const float* hrow = HDN + r * RH;
#pragma unroll
        for (int j = 0; j < 6; ++j) {
            float s = b2[j];
            for (int c = 0; c < RH; ++c) s += hrow[c] * W2[j * RH + c];
            pose[r * 6 + j] = s;
        }
    }
}

// ---------------------------------------------------------------------------
extern "C" void kernel_launch(void* const* d_in, const int* in_sizes, int n_in,
                              void* d_out, int out_size, void* d_ws, size_t ws_size,
                              hipStream_t stream) {
    const float* fv   = (const float*)d_in[0];
    const float* fi   = (const float*)d_in[1];
    // d_in[2] = ts (unused by reference)
    const float* W_ih = (const float*)d_in[3];
    const float* W_hh = (const float*)d_in[4];
    const float* b_ih = (const float*)d_in[5];
    const float* b_hh = (const float*)d_in[6];
    const float* W1   = (const float*)d_in[7];
    const float* b1   = (const float*)d_in[8];
    const float* W2   = (const float*)d_in[9];
    const float* b2   = (const float*)d_in[10];
    float* out = (float*)d_out;
    char*  ws  = (char*)d_ws;

    float* GI    = (float*)(ws + OFF_GI);
    bf16*  XBF   = (bf16*)(ws + OFF_XBF);
    bf16*  YBF   = (bf16*)(ws + OFF_YBF);
    bf16*  WIHBF = (bf16*)(ws + OFF_WIH);
    bf16*  WHHBF = (bf16*)(ws + OFF_WHH);
    bf16*  W1BF  = (bf16*)(ws + OFF_W1);
    float* Hbuf  = (float*)(ws + OFF_H);
    bf16*  HBF   = (bf16*)(ws + OFF_HBF);
    float* GH    = (float*)(ws + OFF_GH);
    float* HDN   = (float*)(ws + OFF_HDN);
    unsigned* bar = (unsigned*)(ws + OFF_BAR);

    float* pose_out = out;                  // [B,S,6]
    float* hn_out   = out + (size_t)BS * 6; // [2,B,H]

    prep_kernel<<<2048, 256, 0, stream>>>(fv, fi, W_ih, W_hh, W1,
                                          XBF, WIHBF, WHHBF, W1BF, bar);

    const unsigned SEQ_BLOCKS = 64;

    // ---- layer 0 ----
    gemm_gi_kernel<<<1024, 256, 0, stream>>>(XBF, WIHBF, GI);
    gru_seq_kernel<<<SEQ_BLOCKS, 256, 0, stream>>>(GI, WHHBF, b_ih, b_hh,
                                                   Hbuf, HBF, GH, YBF,
                                                   hn_out, bar, bar + 1, SEQ_BLOCKS);
    // ---- layer 1 (input = layer-0 outputs; Y1 reuses XBF) ----
    gemm_gi_kernel<<<1024, 256, 0, stream>>>(YBF, WIHBF + (size_t)H3 * H, GI);
    gru_seq_kernel<<<SEQ_BLOCKS, 256, 0, stream>>>(GI, WHHBF + (size_t)H3 * H,
                                                   b_ih + H3, b_hh + H3,
                                                   Hbuf, HBF, GH, XBF,
                                                   hn_out + (size_t)BB * H,
                                                   bar, bar + 1, SEQ_BLOCKS);
    // ---- regressor ----
    regr1_kernel<<<256, 256, 0, stream>>>(XBF, W1BF, b1, HDN);
    regr2_kernel<<<64, 256, 0, stream>>>(HDN, W2, b2, pose_out);
}
